// TinyTransformer_89464168775748
// MI455X (gfx1250) — matrix operand order repaired and checked by reading, not compiled, not run on hardware
//
#include <hip/hip_runtime.h>

// ---------------------------------------------------------------------------
// TinyTransformer forward for gfx1250 (MI455X)
// wave32, v_wmma_f32_16x16x32_f16, TDM (tensor_load_to_lds) staged vocab GEMM
// ---------------------------------------------------------------------------

typedef _Float16 half8   __attribute__((ext_vector_type(8)));
typedef _Float16 half16  __attribute__((ext_vector_type(16)));
typedef float    float8  __attribute__((ext_vector_type(8)));
typedef unsigned int uint4v __attribute__((ext_vector_type(4)));
typedef int      int4v   __attribute__((ext_vector_type(4)));
typedef int      int8v   __attribute__((ext_vector_type(8)));

#define BATCH 4
#define SEQ   2048
#define DIM   512
#define VOCAB 32000
#define QKV3  1536
#define MTOT  8192   // BATCH*SEQ

#if defined(__has_builtin)
#if __has_builtin(__builtin_amdgcn_tensor_load_to_lds) && __has_builtin(__builtin_amdgcn_s_wait_tensorcnt)
#define HAVE_TDM 1
#endif
#endif

#define WMMA_F16(a, b, c) \
  __builtin_amdgcn_wmma_f32_16x16x32_f16(false, (a), false, (b), (short)0, (c), false, false)

// A fragment (16x32 f16, row-major, ld=lda): lanes 0-15 K=0..15,
// lanes 16-31 K=16..31 contiguous.
__device__ __forceinline__ half16 ld_a_frag(const _Float16* __restrict__ p, int lda, int lane) {
  const int r  = lane & 15;
  const int kb = (lane >> 4) << 4;
  const _Float16* q = p + (size_t)r * lda + kb;
  half16 a;
  ((half8*)&a)[0] = *(const half8*)(q);
  ((half8*)&a)[1] = *(const half8*)(q + 8);
  return a;
}

// B fragment (32x16 f16) from B^T row-major [N x K], ld=ldb: lane col = lane&15,
// K = (lane>>4)*16 + 0..15 contiguous.
__device__ __forceinline__ half16 ld_b_frag(const _Float16* __restrict__ p, int ldb, int lane) {
  const int n  = lane & 15;
  const int kb = (lane >> 4) << 4;
  const _Float16* q = p + (size_t)n * ldb + kb;
  half16 b;
  ((half8*)&b)[0] = *(const half8*)(q);
  ((half8*)&b)[1] = *(const half8*)(q + 8);
  return b;
}

// Register double-buffered K-loop: 16(M) x 64(N) per wave, 4 accumulators.
// Loads for step kk+32 are issued before the 4 WMMAs of step kk.
__device__ __forceinline__ void gemm_k_loop(const _Float16* __restrict__ Ab, int lda,
                                            const _Float16* __restrict__ Bb, int ldb,
                                            int kmax, int lane, float8 (&acc)[4]) {
  half16 a  = ld_a_frag(Ab, lda, lane);
  half16 b0 = ld_b_frag(Bb + (size_t)0  * ldb, ldb, lane);
  half16 b1 = ld_b_frag(Bb + (size_t)16 * ldb, ldb, lane);
  half16 b2 = ld_b_frag(Bb + (size_t)32 * ldb, ldb, lane);
  half16 b3 = ld_b_frag(Bb + (size_t)48 * ldb, ldb, lane);
  for (int kk = 0; kk < kmax; kk += 32) {
    const int kn = (kk + 32 < kmax) ? (kk + 32) : kk;   // clamped (always valid) prefetch
    half16 an  = ld_a_frag(Ab + kn, lda, lane);
    half16 bn0 = ld_b_frag(Bb + (size_t)0  * ldb + kn, ldb, lane);
    half16 bn1 = ld_b_frag(Bb + (size_t)16 * ldb + kn, ldb, lane);
    half16 bn2 = ld_b_frag(Bb + (size_t)32 * ldb + kn, ldb, lane);
    half16 bn3 = ld_b_frag(Bb + (size_t)48 * ldb + kn, ldb, lane);
    acc[0] = WMMA_F16(a, b0, acc[0]);
    acc[1] = WMMA_F16(a, b1, acc[1]);
    acc[2] = WMMA_F16(a, b2, acc[2]);
    acc[3] = WMMA_F16(a, b3, acc[3]);
    a = an; b0 = bn0; b1 = bn1; b2 = bn2; b3 = bn3;
  }
}

// -------------------------- 1) embedding gather ----------------------------
__global__ __launch_bounds__(256) void k_gather(const int* __restrict__ x,
                                                const float* __restrict__ embed,
                                                _Float16* __restrict__ h16) {
  size_t i8 = ((size_t)blockIdx.x * 256 + threadIdx.x) * 8;
  int row = (int)(i8 >> 9);
  int col = (int)(i8 & 511);
  int tok = x[row];
  const float* src = embed + (size_t)tok * DIM + col;
  _Float16* dst = h16 + i8;
#pragma unroll
  for (int j = 0; j < 8; ++j) dst[j] = (_Float16)src[j];
}

// ----------------- 2) f32 [K x N] -> f16 transpose [N x K] -----------------
__global__ __launch_bounds__(256) void k_transpose_cvt(const float* __restrict__ in,
                                                       _Float16* __restrict__ out,
                                                       int Kd, int Nd) {
  __shared__ float tile[32][33];
  int n0 = blockIdx.x * 32, k0 = blockIdx.y * 32;
  int tx = threadIdx.x, ty = threadIdx.y;
#pragma unroll
  for (int j = 0; j < 4; ++j)
    tile[ty + j * 8][tx] = in[(size_t)(k0 + ty + j * 8) * Nd + n0 + tx];
  __syncthreads();
#pragma unroll
  for (int j = 0; j < 4; ++j)
    out[(size_t)(n0 + ty + j * 8) * Kd + k0 + tx] = (_Float16)tile[tx][ty + j * 8];
}

// --------------------- 3) QKV GEMM: [8192x512]x[512x1536] ------------------
__global__ __launch_bounds__(256) void k_gemm_qkv(const _Float16* __restrict__ A,
                                                  const _Float16* __restrict__ Bt,
                                                  const float* __restrict__ bias,
                                                  _Float16* __restrict__ Qo,
                                                  _Float16* __restrict__ Ko,
                                                  _Float16* __restrict__ Vt) {
  const int lane = threadIdx.x & 31, wave = threadIdx.x >> 5;
  const int m0 = blockIdx.y * 64 + (wave >> 1) * 16;
  const int n0 = blockIdx.x * 128 + (wave & 1) * 64;
  float8 acc[4] = {};
  gemm_k_loop(A + (size_t)m0 * DIM, DIM, Bt + (size_t)n0 * DIM, DIM, DIM, lane, acc);
  const float qscale = 0.044194173824159216f;  // 1/sqrt(512)
  const int hi8 = (lane >> 4) << 3;
#pragma unroll
  for (int t = 0; t < 4; ++t) {
    int n = n0 + t * 16 + (lane & 15);
    float bv = bias[n];
#pragma unroll
    for (int r = 0; r < 8; ++r) {
      int m = m0 + r + hi8;
      float v = acc[t][r] + bv;
      if (n < DIM) {
        Qo[(size_t)m * DIM + n] = (_Float16)(v * qscale);
      } else if (n < 2 * DIM) {
        Ko[(size_t)m * DIM + (n - DIM)] = (_Float16)v;
      } else {
        int b = m >> 11, s = m & (SEQ - 1);
        Vt[((size_t)b * DIM + (n - 2 * DIM)) * SEQ + s] = (_Float16)v;
      }
    }
  }
}

// ------------------- 4) scores = Q K^T, causal, f32 out --------------------
__global__ __launch_bounds__(256) void k_scores(const _Float16* __restrict__ Q,
                                                const _Float16* __restrict__ K,
                                                float* __restrict__ S) {
  if (blockIdx.x * 128 > blockIdx.y * 64 + 63) return;   // fully masked block
  const int lane = threadIdx.x & 31, wave = threadIdx.x >> 5;
  const int b = blockIdx.z;
  const int m0 = blockIdx.y * 64 + (wave >> 1) * 16;
  const int n0 = blockIdx.x * 128 + (wave & 1) * 64;
  if (n0 > m0 + 15) return;                              // fully masked wave (uniform)
  float8 acc[4] = {};
  gemm_k_loop(Q + (size_t)(b * SEQ + m0) * DIM, DIM,
              K + (size_t)(b * SEQ + n0) * DIM, DIM, DIM, lane, acc);
  float* out = S + (size_t)b * SEQ * SEQ;
  const int hi8 = (lane >> 4) << 3;
#pragma unroll
  for (int t = 0; t < 4; ++t) {
    int n = n0 + t * 16 + (lane & 15);
#pragma unroll
    for (int r = 0; r < 8; ++r) {
      int m = m0 + r + hi8;
      if (n <= m) out[(size_t)m * SEQ + n] = acc[t][r];
    }
  }
}

// ----------------------------- 5) row softmax ------------------------------
__global__ __launch_bounds__(256) void k_softmax(const float* __restrict__ S,
                                                 _Float16* __restrict__ P) {
  __shared__ float red[256];
  const int tid = threadIdx.x;
  const int q = blockIdx.x & (SEQ - 1);
  const float* row = S + (size_t)blockIdx.x * SEQ;
  _Float16* prow = P + (size_t)blockIdx.x * SEQ;
  const int len = q + 1;
  float mx = -3.4e38f;
  for (int i = tid; i < len; i += 256) mx = fmaxf(mx, row[i]);
  red[tid] = mx; __syncthreads();
  for (int s = 128; s > 0; s >>= 1) {
    if (tid < s) red[tid] = fmaxf(red[tid], red[tid + s]);
    __syncthreads();
  }
  mx = red[0]; __syncthreads();
  float sum = 0.f;
  for (int i = tid; i < len; i += 256) sum += __expf(row[i] - mx);
  red[tid] = sum; __syncthreads();
  for (int s = 128; s > 0; s >>= 1) {
    if (tid < s) red[tid] += red[tid + s];
    __syncthreads();
  }
  float inv = 1.0f / red[0];
  for (int i = tid; i < len; i += 256) prow[i] = (_Float16)(__expf(row[i] - mx) * inv);
  const int kend = (((q & ~15) + 16) + 31) & ~31;        // 32-aligned causal bound
  for (int i = len + tid; i < kend; i += 256) prow[i] = (_Float16)0.f;
}

// ------------------- 6) out = P @ V  (K-loop causally bounded) -------------
__global__ __launch_bounds__(256) void k_attn_v(const _Float16* __restrict__ P,
                                                const _Float16* __restrict__ Vt,
                                                _Float16* __restrict__ O) {
  const int lane = threadIdx.x & 31, wave = threadIdx.x >> 5;
  const int b = blockIdx.z;
  const int m0 = blockIdx.y * 64 + (wave >> 1) * 16;
  const int n0 = blockIdx.x * 128 + (wave & 1) * 64;
  const int kend = ((m0 + 16) + 31) & ~31;
  float8 acc[4] = {};
  gemm_k_loop(P + (size_t)b * SEQ * SEQ + (size_t)m0 * SEQ, SEQ,
              Vt + (size_t)b * DIM * SEQ + (size_t)n0 * SEQ, SEQ, kend, lane, acc);
  const int hi8 = (lane >> 4) << 3;
#pragma unroll
  for (int t = 0; t < 4; ++t) {
    int n = n0 + t * 16 + (lane & 15);
#pragma unroll
    for (int r = 0; r < 8; ++r) {
      int m = m0 + r + hi8;
      O[(size_t)(b * SEQ + m) * DIM + n] = (_Float16)acc[t][r];
    }
  }
}

// ------------- 7) logits = out @ fc_w + fc_b : [8192x512]x[512x32000] ------
// TDM-staged B tiles: one tensor_load_to_lds per K-step moves the whole
// 128x32 f16 tile (8KB) into LDS, double-buffered, rows padded 64B->80B so the
// 16-lane B-fragment reads hit 16 distinct banks.
#define BROW 40   // padded LDS row stride in halves (64B data + 16B pad)

#if HAVE_TDM
__device__ __forceinline__ void tdm_load_b_tile(const _Float16* gsrc, unsigned lds_addr) {
  unsigned long long ga = (unsigned long long)(size_t)gsrc;
  uint4v g0 = { 1u,                                   // count=1 valid descriptor
                lds_addr,                             // LDS byte address
                (unsigned)ga,                         // global_addr[31:0]
                ((unsigned)(ga >> 32) & 0x01ffffffu) | (2u << 30) };  // [56:32] | type=2
  int8v g1 = { (int)((1u << 16)        // data_size = 2 bytes
                   | (1u << 20)        // pad_enable
                   | (3u << 22)        // pad_interval: every 16 dwords (one 64B row)
                   | (3u << 25)),      // pad_amount: 4 dwords (16B)
               (int)(512u << 16),      // tensor_dim0 = 512 (row length, elems)
               (int)(32000u << 16),    // tensor_dim1 = 32000 (rows)
               (int)(32u << 16),       // tile_dim0 = 32 elems (one K-step)
               128,                    // tile_dim1 = 128 rows
               512,                    // tensor_dim0_stride = 512 elems
               0, 0 };
  int4v z4 = { 0, 0, 0, 0 };
  int8v z8 = { 0, 0, 0, 0, 0, 0, 0, 0 };
  // 6-arg form (clang-23 / therock-10.0): (g0, g1, g2, g3, g4, cpol)
  __builtin_amdgcn_tensor_load_to_lds(g0, g1, z4, z4, z8, 0);
}
#endif

__global__ __launch_bounds__(256) void k_logits(const _Float16* __restrict__ A,
                                                const _Float16* __restrict__ Bt,
                                                const float* __restrict__ bias,
                                                float* __restrict__ C) {
  const int lane = threadIdx.x & 31, wave = threadIdx.x >> 5;
  const int m0 = blockIdx.y * 64 + (wave >> 1) * 16;
  const int n0wg = blockIdx.x * 128;
  const int n0l = (wave & 1) * 64;
  const _Float16* Ab = A + (size_t)m0 * DIM;
  float8 acc[4] = {};
#if HAVE_TDM
  __shared__ _Float16 bsh[2][128 * BROW];   // 20KB double-buffered B tile
  const _Float16* gB = Bt + (size_t)n0wg * DIM;
  if (wave == 0) tdm_load_b_tile(gB, (unsigned)(size_t)&bsh[0][0]);
  half16 a = ld_a_frag(Ab, DIM, lane);
  for (int kk = 0; kk < DIM; kk += 32) {
    const int cur = (kk >> 5) & 1;
    if (wave == 0) __builtin_amdgcn_s_wait_tensorcnt(0);   // DMA(kk) complete
    __syncthreads();                                       // hand-off + reader drain
    if (wave == 0 && kk + 32 < DIM)
      tdm_load_b_tile(gB + kk + 32, (unsigned)(size_t)&bsh[cur ^ 1][0]);
    const int kn = (kk + 32 < DIM) ? (kk + 32) : kk;
    half16 an = ld_a_frag(Ab + kn, DIM, lane);             // overlap A with WMMA
#pragma unroll
    for (int t = 0; t < 4; ++t) {
      const _Float16* bp = &bsh[cur][(n0l + t * 16 + (lane & 15)) * BROW + ((lane >> 4) << 4)];
      half16 bf;
      ((half8*)&bf)[0] = *(const half8*)(bp);
      ((half8*)&bf)[1] = *(const half8*)(bp + 8);
      acc[t] = WMMA_F16(a, bf, acc[t]);
    }
    a = an;
  }
#else
  gemm_k_loop(Ab, DIM, Bt + (size_t)(n0wg + n0l) * DIM, DIM, DIM, lane, acc);
#endif
  const int hi8 = (lane >> 4) << 3;
#pragma unroll
  for (int t = 0; t < 4; ++t) {
    int n = n0wg + n0l + t * 16 + (lane & 15);
    float bv = bias[n];
#pragma unroll
    for (int r = 0; r < 8; ++r) {
      int m = m0 + r + hi8;
      C[(size_t)m * VOCAB + n] = acc[t][r] + bv;
    }
  }
}

// ---------------------------------------------------------------------------
extern "C" void kernel_launch(void* const* d_in, const int* in_sizes, int n_in,
                              void* d_out, int out_size, void* d_ws, size_t ws_size,
                              hipStream_t stream) {
  (void)in_sizes; (void)n_in; (void)out_size; (void)ws_size;
  const int*   x      = (const int*)d_in[0];
  const float* embed  = (const float*)d_in[1];
  const float* qkv_w  = (const float*)d_in[2];
  const float* qkv_b  = (const float*)d_in[3];
  const float* fc_w   = (const float*)d_in[4];
  const float* fc_b   = (const float*)d_in[5];
  float* logits = (float*)d_out;

  char* ws = (char*)d_ws;
  _Float16* h16   = (_Float16*)(ws);                               //  8,388,608
  _Float16* qkvT  = (_Float16*)(ws + 8388608);                     //  1,572,864
  _Float16* fcT   = (_Float16*)(ws + 9961472);                     // 32,768,000
  _Float16* Q16   = (_Float16*)(ws + 42729472);                    //  8,388,608
  _Float16* K16   = (_Float16*)(ws + 51118080);                    //  8,388,608
  _Float16* V16t  = (_Float16*)(ws + 59506688);                    //  8,388,608  [b][d][s]
  float*    scor  = (float*)   (ws + 67895296);                    // 67,108,864  [b][q][k]
  _Float16* P16   = (_Float16*)(ws + 135004160);                   // 33,554,432
  _Float16* O16   = (_Float16*)(ws + 168558592);                   //  8,388,608

  k_gather<<<(MTOT * DIM) / (256 * 8), 256, 0, stream>>>(x, embed, h16);
  k_transpose_cvt<<<dim3(QKV3 / 32, DIM / 32), dim3(32, 8), 0, stream>>>(qkv_w, qkvT, DIM, QKV3);
  k_transpose_cvt<<<dim3(VOCAB / 32, DIM / 32), dim3(32, 8), 0, stream>>>(fc_w, fcT, DIM, VOCAB);
  k_gemm_qkv<<<dim3(QKV3 / 128, MTOT / 64), 256, 0, stream>>>(h16, qkvT, qkv_b, Q16, K16, V16t);
  k_scores<<<dim3(SEQ / 128, SEQ / 64, BATCH), 256, 0, stream>>>(Q16, K16, scor);
  k_softmax<<<BATCH * SEQ, 256, 0, stream>>>(scor, P16);
  k_attn_v<<<dim3(DIM / 128, SEQ / 64, BATCH), 256, 0, stream>>>(P16, V16t, O16);
  k_logits<<<dim3(VOCAB / 128, MTOT / 64), 256, 0, stream>>>(O16, fcT, fc_b, logits);
}
